// CrossDecoderLayer_30992484008302
// MI455X (gfx1250) — matrix operand order repaired and checked by reading, not compiled, not run on hardware
//
#include <hip/hip_runtime.h>
#include <hip/hip_bf16.h>
#include <math.h>

#define D_  512
#define H_  8
#define DK_ 64
#define BS_ 8
#define NW_ 256
#define NS_ 32
#define NV_ 64
#define NC_ 128

typedef __attribute__((ext_vector_type(16))) _Float16 v16h;
typedef __attribute__((ext_vector_type(8)))  _Float16 v8h;
typedef __attribute__((ext_vector_type(8)))  float    v8f;
typedef __attribute__((ext_vector_type(4)))  unsigned int v4u;
typedef __attribute__((ext_vector_type(8)))  int      v8i;
typedef __attribute__((ext_vector_type(4)))  int      v4i;

__device__ __forceinline__ float4 ld4(const float* p) {
    return *(const float4*)p;        // 16-byte aligned by construction
}

__device__ __forceinline__ v16h pack16(float4 a0, float4 a1, float4 a2, float4 a3) {
    v16h f;
    f[0]=(_Float16)a0.x; f[1]=(_Float16)a0.y; f[2]=(_Float16)a0.z; f[3]=(_Float16)a0.w;
    f[4]=(_Float16)a1.x; f[5]=(_Float16)a1.y; f[6]=(_Float16)a1.z; f[7]=(_Float16)a1.w;
    f[8]=(_Float16)a2.x; f[9]=(_Float16)a2.y; f[10]=(_Float16)a2.z; f[11]=(_Float16)a2.w;
    f[12]=(_Float16)a3.x; f[13]=(_Float16)a3.y; f[14]=(_Float16)a3.z; f[15]=(_Float16)a3.w;
    return f;
}

__device__ __forceinline__ v8h pack8(float4 a, float4 b) {
    v8h f;
    f[0]=(_Float16)a.x; f[1]=(_Float16)a.y; f[2]=(_Float16)a.z; f[3]=(_Float16)a.w;
    f[4]=(_Float16)b.x; f[5]=(_Float16)b.y; f[6]=(_Float16)b.z; f[7]=(_Float16)b.w;
    return f;
}

// ---------------------------------------------------------------------------
// Tensor Data Mover: DMA a rows x cols f32 tile (row stride in elements) from
// global memory into LDS at byte offset lds_off.  D# built per ISA 08 ch.8.
// Tracked by TENSORcnt.
// ---------------------------------------------------------------------------
__device__ __forceinline__ void tdm_load_tile(unsigned lds_off, const float* gptr,
                                              int rows, int cols, long row_stride)
{
    const unsigned long long ga = (unsigned long long)(size_t)gptr;
    v4u g0;
    g0[0] = 1u;                                          // count=1, user mode
    g0[1] = lds_off;                                     // LDS byte address
    g0[2] = (unsigned)(ga & 0xFFFFFFFFu);                // global_addr[31:0]
    g0[3] = (unsigned)((ga >> 32) & 0x01FFFFFFu) | (2u << 30);   // addr[56:32], type=2
    const unsigned td0 = (unsigned)cols;
    const unsigned td1 = (unsigned)rows;
    const unsigned long long st0 = (unsigned long long)row_stride;
    v8i g1;
    g1[0] = (int)(2u << 16);                             // data_size = 4 bytes
    g1[1] = (int)((td0 & 0xFFFFu) << 16);                // tensor_dim0[15:0]
    g1[2] = (int)((td0 >> 16) | ((td1 & 0xFFFFu) << 16));// tensor_dim0[31:16] | dim1[15:0]
    g1[3] = (int)((td1 >> 16) | (td0 << 16));            // dim1[31:16] | tile_dim0
    g1[4] = (int)(td1 & 0xFFFFu);                        // tile_dim1 (tile_dim2 = 0)
    g1[5] = (int)(st0 & 0xFFFFFFFFu);                    // tensor_dim0_stride[31:0]
    g1[6] = (int)((st0 >> 32) & 0xFFFFu);                // stride[47:32]; dim1_stride=0
    g1[7] = 0;
    const v4i z4 = {0, 0, 0, 0};
#if __has_include(<hip/amd_detail/amd_gfx1250_TDM.h>)
    const v8i z8 = {0, 0, 0, 0, 0, 0, 0, 0};
    __builtin_amdgcn_tensor_load_to_lds(g0, g1, z4, z4, z8, 0);
#else
    __builtin_amdgcn_tensor_load_to_lds(g0, g1, z4, z4, 0);
#endif
}

// ---------------------------------------------------------------------------
// TDM-staged WMMA GEMM for plain linears (C = A * W^T + bias [+resid][relu]).
// Block = 8 waves = 128x64 output tile sharing one 64-column B panel.
// Wave 0 DMAs 64x64 f32 weight chunks into a double-buffered LDS staging
// buffer via tensor_load_to_lds; the block then converts the panel to f16
// ONCE cooperatively, and the WMMA loop reads B fragments as 2x ds_load_b128
// of f16 with no per-wave conversions.
// ---------------------------------------------------------------------------
__global__ void __launch_bounds__(256)
gemm_wmma_tdm(const float* __restrict__ A, int lam,
              const float* __restrict__ Bt, int lbn,
              float* __restrict__ C, int ldc,
              const float* __restrict__ bias,
              const float* __restrict__ resid,
              int M, int N, int K, int do_relu)
{
    __shared__ float4 Bs4[2][16 * 64];                  // f32 staging (TDM dest)
    __shared__ v8h    Bh[2][8 * 64];                    // f16 panel (64 rows x 64)
    const int wave = threadIdx.x >> 5;
    const int lane = threadIdx.x & 31;
    const int n64  = N >> 6;
    const int m0   = (blockIdx.x / n64) * 128 + wave * 16;
    const int n0   = (blockIdx.x % n64) * 64;
    const int half = lane >> 4;
    const int quad = lane & 15;

    const float* Arow  = A  + (long)(m0 + quad) * lam;
    const float* Bbase = Bt + (long)n0 * lbn;

    v8f acc0 = {0.f,0.f,0.f,0.f,0.f,0.f,0.f,0.f};
    v8f acc1 = acc0, acc2 = acc0, acc3 = acc0;

    const int nk = K >> 6;                              // 64-wide K chunks
    if (wave == 0)
        tdm_load_tile((unsigned)(size_t)&Bs4[0][0], Bbase, 64, 64, lbn);

    for (int kc = 0; kc < nk; ++kc) {
        if (wave == 0) {
            if (kc + 1 < nk) {
                tdm_load_tile((unsigned)(size_t)&Bs4[(kc + 1) & 1][0],
                              Bbase + (long)(kc + 1) * 64, 64, 64, lbn);
                __builtin_amdgcn_s_wait_tensorcnt(1);   // current chunk landed
            } else {
                __builtin_amdgcn_s_wait_tensorcnt(0);
            }
        }
        __syncthreads();                                // staging kc visible

        // Cooperative one-shot f32 -> f16 panel conversion (16 floats/thread).
        {
            const float4* src = &Bs4[kc & 1][0];
            v8h*          dst = &Bh[kc & 1][0];
            const int t4 = threadIdx.x * 4;
            const float4 x0 = src[t4 + 0];
            const float4 x1 = src[t4 + 1];
            const float4 x2 = src[t4 + 2];
            const float4 x3 = src[t4 + 3];
            dst[threadIdx.x * 2 + 0] = pack8(x0, x1);
            dst[threadIdx.x * 2 + 1] = pack8(x2, x3);
        }
        __syncthreads();                                // f16 panel ready

        const v8h* Bbuf = &Bh[kc & 1][0];
        const int kg = kc * 64;
#pragma unroll
        for (int s = 0; s < 2; ++s) {
            const int k0 = kg + s * 32;
            __builtin_prefetch(Arow + k0 + 128, 0, 1);
            const v16h af = pack16(ld4(Arow + k0 + 8 * half),
                                   ld4(Arow + k0 + 8 * half + 4),
                                   ld4(Arow + k0 + 16 + 8 * half),
                                   ld4(Arow + k0 + 16 + 8 * half + 4));
            v8f* accs[4] = {&acc0, &acc1, &acc2, &acc3};
#pragma unroll
            for (int j = 0; j < 4; ++j) {
                const int base = (16 * j + quad) * 8 + s * 4 + half;
                const v8h lo = Bbuf[base + 0];
                const v8h hi = Bbuf[base + 2];
                const v16h bf = __builtin_shufflevector(lo, hi,
                        0,1,2,3,4,5,6,7,8,9,10,11,12,13,14,15);
                *accs[j] = __builtin_amdgcn_wmma_f32_16x16x32_f16(false, af, false, bf,
                                                                  (short)0, *accs[j], false, false);
            }
        }
        // next iteration's first barrier gates staging overwrite; the f16
        // panel buffers alternate, so no extra barrier is needed here.
    }

    v8f accv[4] = {acc0, acc1, acc2, acc3};
#pragma unroll
    for (int j = 0; j < 4; ++j) {
        const int   n  = n0 + 16 * j + quad;
        const float bv = bias ? bias[n] : 0.f;
#pragma unroll
        for (int r = 0; r < 8; ++r) {
            const int m = m0 + half * 8 + r;
            float v = accv[j][r] + bv;
            if (resid)   v += resid[(long)m * ldc + n];
            if (do_relu) v  = fmaxf(v, 0.f);
            C[(long)m * ldc + n] = v;
        }
    }
}

// ---------------------------------------------------------------------------
// Fast-path WMMA GEMM (K contiguous, batched) — used for Q*K^T.
// ---------------------------------------------------------------------------
__global__ void __launch_bounds__(256)
gemm_wmma_fast(const float* __restrict__ A, long sAz, long sAy, int lam,
               const float* __restrict__ Bt, long sBz, long sBy, int lbn,
               float* __restrict__ C, long sCz, long sCy, int ldc,
               int M, int N, int K)
{
    A  += (long)blockIdx.z * sAz + (long)blockIdx.y * sAy;
    Bt += (long)blockIdx.z * sBz + (long)blockIdx.y * sBy;
    C  += (long)blockIdx.z * sCz + (long)blockIdx.y * sCy;

    const int wave = threadIdx.x >> 5;
    const int lane = threadIdx.x & 31;
    const int n64  = N >> 6;
    const int tile = blockIdx.x * 8 + wave;
    if (tile >= (M >> 4) * n64) return;

    const int m0   = (tile / n64) << 4;
    const int n0   = (tile % n64) << 6;
    const int half = lane >> 4;
    const int quad = lane & 15;

    const float* Arow  = A + (long)(m0 + quad) * lam;
    const float* Brow0 = Bt + (long)(n0 + quad) * lbn;
    const float* Brow1 = Brow0 + (long)16 * lbn;
    const float* Brow2 = Brow0 + (long)32 * lbn;
    const float* Brow3 = Brow0 + (long)48 * lbn;

    v8f acc0 = {0.f,0.f,0.f,0.f,0.f,0.f,0.f,0.f};
    v8f acc1 = acc0, acc2 = acc0, acc3 = acc0;

    for (int k0 = 0; k0 < K; k0 += 32) {
        __builtin_prefetch(Arow + k0 + 64, 0, 1);
        __builtin_prefetch(Brow0 + k0 + 64, 0, 1);
        const v16h af = pack16(ld4(Arow + k0 + 16 * half),
                               ld4(Arow + k0 + 16 * half + 4),
                               ld4(Arow + k0 + 16 * half + 8),
                               ld4(Arow + k0 + 16 * half + 12));
        const float* brows[4] = {Brow0, Brow1, Brow2, Brow3};
        v8f* accs[4] = {&acc0, &acc1, &acc2, &acc3};
#pragma unroll
        for (int j = 0; j < 4; ++j) {
            const float* br = brows[j] + k0 + 16 * half;
            const v16h bf = pack16(ld4(br), ld4(br + 4), ld4(br + 8), ld4(br + 12));
            *accs[j] = __builtin_amdgcn_wmma_f32_16x16x32_f16(false, af, false, bf,
                                                              (short)0, *accs[j], false, false);
        }
    }

    v8f accv[4] = {acc0, acc1, acc2, acc3};
#pragma unroll
    for (int j = 0; j < 4; ++j) {
        const int n = n0 + 16 * j + quad;
#pragma unroll
        for (int r = 0; r < 8; ++r) {
            const int m = m0 + half * 8 + r;
            C[(long)m * ldc + n] = accv[j][r];
        }
    }
}

// ---------------------------------------------------------------------------
// Generic strided WMMA GEMM — only for the small attention A*V einsums.
// ---------------------------------------------------------------------------
__global__ void __launch_bounds__(256)
gemm_wmma_kernel(const float* __restrict__ A, long sAz, long sAy, int lam, int lak,
                 const float* __restrict__ Bt, long sBz, long sBy, int lbn, int lbk,
                 float* __restrict__ C, long sCz, long sCy, int ldc,
                 int M, int N, int K)
{
    A  += (long)blockIdx.z * sAz + (long)blockIdx.y * sAy;
    Bt += (long)blockIdx.z * sBz + (long)blockIdx.y * sBy;
    C  += (long)blockIdx.z * sCz + (long)blockIdx.y * sCy;

    const int wave   = threadIdx.x >> 5;
    const int lane   = threadIdx.x & 31;
    const int ntiles = N >> 4;
    const int tile   = blockIdx.x * 8 + wave;
    if (tile >= (M >> 4) * ntiles) return;

    const int m0   = (tile / ntiles) << 4;
    const int n0   = (tile % ntiles) << 4;
    const int half = lane >> 4;
    const int quad = lane & 15;

    const float* Arow = A  + (long)(m0 + quad) * lam;
    const float* Brow = Bt + (long)(n0 + quad) * lbn;

    v8f acc = {0.f,0.f,0.f,0.f,0.f,0.f,0.f,0.f};
    for (int k0 = 0; k0 < K; k0 += 32) {
        v16h a, b;
#pragma unroll
        for (int e = 0; e < 16; ++e) {
            const int ka = k0 + 8 * half + e + ((e >= 8) ? 8 : 0);
            a[e] = (_Float16)Arow[(long)ka * lak];
        }
#pragma unroll
        for (int e = 0; e < 16; ++e) {
            const int kb = k0 + 8 * half + e + ((e >= 8) ? 8 : 0);
            b[e] = (_Float16)Brow[(long)kb * lbk];
        }
        acc = __builtin_amdgcn_wmma_f32_16x16x32_f16(false, a, false, b,
                                                     (short)0, acc, false, false);
    }
    const int n = n0 + quad;
#pragma unroll
    for (int r = 0; r < 8; ++r) {
        const int m = m0 + half * 8 + r;
        C[(long)m * ldc + n] = acc[r];
    }
}

// ---------------------------------------------------------------------------
__global__ void __launch_bounds__(256)
layernorm_kernel(const float* __restrict__ x, const float* __restrict__ r,
                 const float* __restrict__ g, const float* __restrict__ b,
                 float* __restrict__ y)
{
    const int  wave = threadIdx.x >> 5;
    const int  lane = threadIdx.x & 31;
    const long row  = (long)blockIdx.x * 8 + wave;
    const float* xr = x + row * D_;
    const float* rr = r ? r + row * D_ : nullptr;

    float v[16];
    float s = 0.f;
#pragma unroll
    for (int i = 0; i < 16; ++i) {
        float t = xr[lane + 32 * i];
        if (rr) t += rr[lane + 32 * i];
        v[i] = t; s += t;
    }
#pragma unroll
    for (int o = 16; o > 0; o >>= 1) s += __shfl_xor(s, o, 32);
    const float mean = s * (1.f / D_);
    float vs = 0.f;
#pragma unroll
    for (int i = 0; i < 16; ++i) { const float d = v[i] - mean; vs += d * d; }
#pragma unroll
    for (int o = 16; o > 0; o >>= 1) vs += __shfl_xor(vs, o, 32);
    const float inv = rsqrtf(vs * (1.f / D_) + 1e-5f);
#pragma unroll
    for (int i = 0; i < 16; ++i) {
        const int c = lane + 32 * i;
        y[row * D_ + c] = (v[i] - mean) * inv * g[c] + b[c];
    }
}

// ---------------------------------------------------------------------------
__global__ void __launch_bounds__(256)
softmax_mask_kernel(float* __restrict__ s, const unsigned char* __restrict__ mask,
                    int Q, int Kl, float scale)
{
    const int  wave = threadIdx.x >> 5;
    const int  lane = threadIdx.x & 31;
    const long row  = (long)blockIdx.x * 8 + wave;
    const long b    = row / ((long)H_ * Q);
    const int  q    = (int)(row % Q);
    float* sr = s + row * (long)Kl;
    const unsigned char* mr = mask + (b * Q + q) * (long)Kl;

    const int nv = Kl >> 5;
    float v[8];
    float mx = -3.0e38f;
    for (int i = 0; i < nv; ++i) {
        const int c = lane + 32 * i;
        const float t = mr[c] ? sr[c] * scale : -1e9f;
        v[i] = t; mx = fmaxf(mx, t);
    }
    for (int o = 16; o > 0; o >>= 1) mx = fmaxf(mx, __shfl_xor(mx, o, 32));
    float sum = 0.f;
    for (int i = 0; i < nv; ++i) { v[i] = expf(v[i] - mx); sum += v[i]; }
    for (int o = 16; o > 0; o >>= 1) sum += __shfl_xor(sum, o, 32);
    const float rs = 1.f / sum;
    for (int i = 0; i < nv; ++i) sr[lane + 32 * i] = v[i] * rs;
}

// ---------------------------------------------------------------------------
__global__ void __launch_bounds__(256)
ca_kernel(const float* __restrict__ qc, const float* __restrict__ kc,
          const float* __restrict__ vc, const int* __restrict__ idx,
          float* __restrict__ out_raw, float* __restrict__ attn_mean)
{
    const long bw = blockIdx.x;
    const int  b  = (int)(bw / NW_);
    const int  t  = threadIdx.x;
    __shared__ float attn_s[NV_ * H_];

    const int sent = idx[bw];
    const float* qrow  = qc + bw * D_;
    const long   avoff = ((long)(b * NS_ + sent)) * NV_ * D_;
    const float* kbase = kc + avoff;
    const float* vbase = vc + avoff;

    for (int p = t; p < NV_ * H_; p += 256) {
        const int n = p >> 3, h = p & 7;
        const float* kr = kbase + (long)n * D_ + h * DK_;
        const float* qh = qrow + h * DK_;
        float s = 0.f;
#pragma unroll
        for (int d = 0; d < DK_; ++d) s += qh[d] * kr[d];
        s *= 0.125f;
        attn_s[p] = 1.f / (1.f + expf(-s));
    }
    __syncthreads();

    for (int c = t; c < D_; c += 256) {
        const int h = c >> 6;
        float acc = 0.f;
        for (int n = 0; n < NV_; ++n) acc += attn_s[n * 8 + h] * vbase[(long)n * D_ + c];
        out_raw[bw * D_ + c] = acc;
    }
    if (t < NV_) {
        float s = 0.f;
#pragma unroll
        for (int h = 0; h < H_; ++h) s += attn_s[t * 8 + h];
        attn_mean[bw * NV_ + t] = s * 0.125f;
    }
}

// ---------------------------------------------------------------------------
__global__ void __launch_bounds__(256)
stack2_kernel(const float* __restrict__ a, const float* __restrict__ b,
              float* __restrict__ out)
{
    const long i   = (long)blockIdx.x * blockDim.x + threadIdx.x;
    const long row = i / D_;
    const int  c   = (int)(i % D_);
    out[(row * 2) * D_ + c]     = a[i];
    out[(row * 2 + 1) * D_ + c] = b[i];
}

// ---------------------------------------------------------------------------
__global__ void __launch_bounds__(256)
fusion_kernel(const float* __restrict__ qf, const float* __restrict__ kf,
              const float* __restrict__ vf, float* __restrict__ out)
{
    const long bw = blockIdx.x;
    const int  t  = threadIdx.x;
    __shared__ float a_s[16];
    __shared__ float p_s[16];

    if (t < 16) {
        const int m = t >> 3, h = t & 7;
        const float* q = qf + bw * D_ + h * DK_;
        const float* k = kf + (bw * 2 + m) * D_ + h * DK_;
        float s = 0.f;
#pragma unroll
        for (int d = 0; d < DK_; ++d) s += q[d] * k[d];
        a_s[m * 8 + h] = s * 0.125f;
    }
    __syncthreads();
    if (t < 8) {
        const float s0 = a_s[t], s1 = a_s[8 + t];
        const float mx = fmaxf(s0, s1);
        const float e0 = expf(s0 - mx), e1 = expf(s1 - mx);
        const float inv = 1.f / (e0 + e1);
        p_s[t] = e0 * inv; p_s[8 + t] = e1 * inv;
    }
    __syncthreads();
    for (int c = t; c < D_; c += 256) {
        const int h = c >> 6;
        out[bw * D_ + c] = p_s[h] * vf[(bw * 2) * D_ + c]
                         + p_s[8 + h] * vf[(bw * 2 + 1) * D_ + c];
    }
}

// ---------------------------------------------------------------------------
// Host-side launch helpers
// ---------------------------------------------------------------------------
static inline void launch_qk(hipStream_t st,
    const float* A, long sAz, long sAy, int lam,
    const float* Bt, long sBz, long sBy, int lbn,
    float* C, long sCz, long sCy, int ldc,
    int M, int N, int K, int gy, int gz)
{
    const int tiles = (M / 16) * (N / 64);
    dim3 grid((tiles + 7) / 8, gy, gz);
    gemm_wmma_fast<<<grid, 256, 0, st>>>(A, sAz, sAy, lam, Bt, sBz, sBy, lbn,
                                         C, sCz, sCy, ldc, M, N, K);
}

static inline void launch_av(hipStream_t st,
    const float* A, long sAz, long sAy, int lam, int lak,
    const float* Bt, long sBz, long sBy, int lbn, int lbk,
    float* C, long sCz, long sCy, int ldc,
    int M, int N, int K, int gy, int gz)
{
    const int tiles = (M / 16) * (N / 16);
    dim3 grid((tiles + 7) / 8, gy, gz);
    gemm_wmma_kernel<<<grid, 256, 0, st>>>(A, sAz, sAy, lam, lak, Bt, sBz, sBy, lbn, lbk,
                                           C, sCz, sCy, ldc, M, N, K);
}

static inline void linear_g(hipStream_t st, const float* X, const float* Wm,
                            const float* b, float* Y, int M, int N, int K,
                            const float* resid = nullptr, int relu = 0)
{
    dim3 grid((M / 128) * (N / 64));
    gemm_wmma_tdm<<<grid, 256, 0, st>>>(X, K, Wm, K, Y, N, b, resid, M, N, K, relu);
}

static inline void ln(hipStream_t st, const float* x, const float* r,
                      const float* g, const float* b, float* y, int rows)
{
    layernorm_kernel<<<rows / 8, 256, 0, st>>>(x, r, g, b, y);
}

extern "C" void kernel_launch(void* const* d_in, const int* in_sizes, int n_in,
                              void* d_out, int out_size, void* d_ws, size_t ws_size,
                              hipStream_t stream)
{
    (void)in_sizes; (void)n_in; (void)out_size; (void)ws_size;
    const float* dialog_in = (const float*)d_in[0];
    const float* map2d     = (const float*)d_in[1];
    const float* caption   = (const float*)d_in[2];
    const unsigned char* dmask = (const unsigned char*)d_in[3];
    const unsigned char* cmask = (const unsigned char*)d_in[4];
    const int* sent_idx    = (const int*)d_in[5];
    const float* const* P  = (const float* const*)(d_in + 6);

    const float *ta_qw=P[0], *ta_qb=P[1], *ta_kw=P[2], *ta_kb=P[3];
    const float *ta_vw=P[4], *ta_vb=P[5], *ta_ow=P[6], *ta_ob=P[7];
    const float *r1g=P[8], *r1b=P[9];
    const float *ca_qw=P[10], *ca_qb=P[11], *ca_kw=P[12], *ca_kb=P[13];
    const float *ca_vw=P[14], *ca_vb=P[15], *ca_ng=P[16], *ca_nb=P[17];
    const float *ca_tg=P[18], *ca_tb=P[19], *ca_ag=P[20], *ca_ab=P[21];
    const float *cng=P[22], *cnb=P[23];
    const float *cp_qw=P[24], *cp_qb=P[25], *cp_kw=P[26], *cp_kb=P[27];
    const float *cp_vw=P[28], *cp_vb=P[29], *cp_ow=P[30], *cp_ob=P[31];
    const float *fu_qw=P[32], *fu_qb=P[33], *fu_kw=P[34], *fu_kb=P[35];
    const float *fu_vw=P[36], *fu_vb=P[37], *fu_mg=P[38], *fu_mb=P[39];
    const float *fu_qng=P[40], *fu_qnb=P[41];
    const float *ng=P[42], *nb=P[43];
    const float *w1=P[44], *b1=P[45], *w2=P[46], *b2=P[47];
    const float *r2g=P[48], *r2b=P[49];

    const long MEG = 1 << 20;
    float* Wk = (float*)d_ws;
    float* W0 = Wk;                 // 1M  scratch
    float* W1 = Wk + 1 * MEG;       // 1M
    float* W2 = Wk + 2 * MEG;       // 1M
    float* W3 = Wk + 3 * MEG;       // 1M
    float* W4 = Wk + 4 * MEG;       // 4M  scores / ffn hidden
    float* W5 = Wk + 8 * MEG;       // 1M  dialog (persists)
    float* W6 = Wk + 9 * MEG;       // 8M  avn -> res2
    float* W7 = Wk + 17 * MEG;      // 8M  kc  -> mod
    float* W8 = Wk + 25 * MEG;      // 8M  vc  -> modn/kf/vf
    float* W9 = Wk + 33 * MEG;      // 1M  ca_out (persists)
    float* out_f    = (float*)d_out;
    float* attn_out = out_f + (long)BS_ * NW_ * D_;

    const long sQ = (long)NW_ * D_;
    // ---- stage 1: pre-norm self attention -------------------------------
    ln(stream, dialog_in, nullptr, r1g, r1b, W0, BS_ * NW_);          // xn
    linear_g(stream, W0, ta_qw, ta_qb, W1, BS_ * NW_, D_, D_);        // q1
    linear_g(stream, W0, ta_kw, ta_kb, W2, BS_ * NW_, D_, D_);        // k1
    linear_g(stream, W0, ta_vw, ta_vb, W3, BS_ * NW_, D_, D_);        // v1
    launch_qk(stream, W1, sQ, DK_, D_,  W2, sQ, DK_, D_,              // Q·K^T
              W4, (long)H_ * NW_ * NW_, (long)NW_ * NW_, NW_, NW_, NW_, DK_, H_, BS_);
    softmax_mask_kernel<<<BS_ * H_ * NW_ / 8, 256, 0, stream>>>(W4, dmask, NW_, NW_, 0.125f);
    launch_av(stream, W4, (long)H_ * NW_ * NW_, (long)NW_ * NW_, NW_, 1,   // A·V
              W3, sQ, DK_, 1, D_,  W0, sQ, DK_, D_, NW_, DK_, NW_, H_, BS_);
    linear_g(stream, W0, ta_ow, ta_ob, W5, BS_ * NW_, D_, D_, dialog_in);  // dialog

    // ---- stage 2: gather-indexed sigmoid cross attention ----------------
    ln(stream, W5, nullptr, ca_tg, ca_tb, W1, BS_ * NW_);             // textn
    ln(stream, map2d, nullptr, ca_ag, ca_ab, W6, BS_ * NS_ * NV_);    // avn
    linear_g(stream, W1, ca_qw, ca_qb, W2, BS_ * NW_, D_, D_);        // qc
    linear_g(stream, W6, ca_kw, ca_kb, W7, BS_ * NS_ * NV_, D_, D_);  // kc (big)
    linear_g(stream, W6, ca_vw, ca_vb, W8, BS_ * NS_ * NV_, D_, D_);  // vc (big)
    ca_kernel<<<BS_ * NW_, 256, 0, stream>>>(W2, W7, W8, sent_idx, W0, attn_out);
    ln(stream, W0, nullptr, ca_ng, ca_nb, W9, BS_ * NW_);             // ca_out

    // ---- stage 3: caption cross attention -------------------------------
    ln(stream, caption, nullptr, cng, cnb, W1, BS_ * NC_);            // capn
    linear_g(stream, W5, cp_qw, cp_qb, W2, BS_ * NW_, D_, D_);        // q2
    linear_g(stream, W1, cp_kw, cp_kb, W3, BS_ * NC_, D_, D_);        // k2
    float* v2 = W3 + (long)BS_ * NC_ * D_;
    linear_g(stream, W1, cp_vw, cp_vb, v2, BS_ * NC_, D_, D_);        // v2
    const long sC = (long)NC_ * D_;
    launch_qk(stream, W2, sQ, DK_, D_,  W3, sC, DK_, D_,              // Q·K^T
              W4, (long)H_ * NW_ * NC_, (long)NW_ * NC_, NC_, NW_, NC_, DK_, H_, BS_);
    softmax_mask_kernel<<<BS_ * H_ * NW_ / 8, 256, 0, stream>>>(W4, cmask, NW_, NC_, 0.125f);
    launch_av(stream, W4, (long)H_ * NW_ * NC_, (long)NW_ * NC_, NC_, 1,   // A·V
              v2, sC, DK_, 1, D_,  W1, sQ, DK_, D_, NW_, DK_, NC_, H_, BS_);
    linear_g(stream, W1, cp_ow, cp_ob, W6, BS_ * NW_, D_, D_);        // res2

    // ---- stage 4: modality fusion ---------------------------------------
    stack2_kernel<<<(BS_ * NW_ * D_) / 256, 256, 0, stream>>>(W9, W6, W7);  // mod
    ln(stream, W7, nullptr, fu_mg, fu_mb, W8, BS_ * NW_ * 2);         // modn
    ln(stream, W5, nullptr, fu_qng, fu_qnb, W0, BS_ * NW_);           // qn
    linear_g(stream, W0, fu_qw, fu_qb, W1, BS_ * NW_, D_, D_);        // qf
    float* kf = W8 + 2 * MEG;
    float* vf = W8 + 4 * MEG;
    linear_g(stream, W8, fu_kw, fu_kb, kf, BS_ * NW_ * 2, D_, D_);    // kf
    linear_g(stream, W8, fu_vw, fu_vb, vf, BS_ * NW_ * 2, D_, D_);    // vf
    fusion_kernel<<<BS_ * NW_, 256, 0, stream>>>(W1, kf, vf, W2);     // resf

    // ---- stage 5: LN + FFN + residual -----------------------------------
    ln(stream, W5, W2, ng, nb, W0, BS_ * NW_);                        // dialog2
    ln(stream, W0, nullptr, r2g, r2b, W1, BS_ * NW_);                 // ffin
    linear_g(stream, W1, w1, b1, W4, BS_ * NW_, 2 * D_, D_, nullptr, 1);   // relu
    linear_g(stream, W4, w2, b2, out_f, BS_ * NW_, D_, 2 * D_, W0);   // final out
}
